// basic_block_15547781612255
// MI455X (gfx1250) — compile-verified, matrix operand order checked by reading, not run on hardware
//
#include <hip/hip_runtime.h>
#include <hip/hip_bf16.h>
#include <math.h>

// ---------------------------------------------------------------------------
// CDNA5 (gfx1250) implementation of the attention+MLP block.
// All GEMMs run on v_wmma_f32_16x16x32_bf16 (wave32), with the shared B-tile
// staged through LDS (async load-to-LDS when the toolchain exposes it).
// ---------------------------------------------------------------------------

typedef __bf16 bf16_t;
typedef __attribute__((ext_vector_type(16))) __bf16 v16bf;
typedef __attribute__((ext_vector_type(8)))  float  v8f;
typedef __attribute__((ext_vector_type(4)))  int    v4i_t;

#if defined(__gfx1250__) && \
    __has_builtin(__builtin_amdgcn_global_load_async_to_lds_b128) && \
    __has_builtin(__builtin_amdgcn_s_wait_asynccnt)
#define ASYNC_LDS 1
#endif

#define B_  8
#define C_  768
#define H_  64
#define W_  64
#define S_  4096       // tokens per image
#define SG_ 256        // global (pooled) tokens per image
#define MG_ 2048       // B_*SG_
#define NTOK_ 32768    // B_*S_
#define HID_ 3072
#define HEADS_ 8
#define HD_ 96
#define NWIN_ 512      // B_ * 8 * 8
#define WS_ 8

static __device__ inline int imin(int a, int b) { return a < b ? a : b; }

// ---- WMMA fragment: 16 bf16 per lane --------------------------------------
// A (16x32, bf16): lanes 0-15 hold row M=lane, K = [0..7]+[16..23];
//                  lanes 16-31 hold row M=lane-16, K = [8..15]+[24..31].
// B (32x16, bf16) mirrors with N in place of M (K-contiguous source rows).
union FragBF {
    v16bf v;
    uint4 q[2];
};

static __device__ inline FragBF load_frag(const bf16_t* base, long row, int K,
                                          int k0, int koff) {
    FragBF f;
    const char* p = (const char*)base + (row * (long)K + (long)(k0 + koff)) * 2;
    f.q[0] = *(const uint4*)p;           // K = k0+koff   .. +7
    f.q[1] = *(const uint4*)(p + 32);    // K = k0+16+koff.. +7
    return f;
}

union AccU { v8f v; float f[8]; };

// ---------------------------------------------------------------------------
// Generic batched GEMM: D[m][n] = act(sum_k A[m][k] * W[n][k] + bias[n])
// A: bf16 (M x K, K-contig), W: bf16 (N x K, K-contig).
// batch = blockIdx.z; bb = batch/HB, hh = batch%HB.
// Out element index = bb*oStrideB + hh*oStrideH + m*oRowStride + n.
// ACT: 0=none, 1=exact GELU. OUTBF: 1=bf16 out, 0=f32 out.
// Block: 256 threads = 8 waves; tile 256(M) x 64(N); K step 32.
// Per wave: 2 M-subtiles x 4 N-subtiles = 8 WMMAs per K step; the 64x32 B
// tile is staged once per block into LDS and fragment-loaded via b128 reads.
// Control flow is block-uniform so EXEC stays all-1s around every WMMA.
// ---------------------------------------------------------------------------
template <int ACT, int OUTBF>
__global__ __launch_bounds__(256) void gemm_wmma_kernel(
    const bf16_t* __restrict__ A, const bf16_t* __restrict__ W,
    const float* __restrict__ bias, char* __restrict__ Out,
    int M, int N, int K,
    long aBatchStride, long wBatchStride, int HB,
    long oStrideB, long oStrideH, int oRowStride)
{
    __shared__ alignas(16) bf16_t ldsB[64 * 32];   // B tile: [n_local][k_local]

    const int batch = blockIdx.z;
    const bf16_t* Ab = A + (long)batch * aBatchStride;
    const bf16_t* Wb = W + (long)batch * wBatchStride;
    const int bb = batch / HB;
    const int hh = batch % HB;
    const long obase = (long)bb * oStrideB + (long)hh * oStrideH;

    const int tid  = threadIdx.x;
    const int wave = tid >> 5;
    const int lane = tid & 31;
    const int r    = lane & 15;
    const int hi   = lane >> 4;
    const int koff = hi << 3;

    const int m0 = blockIdx.y * 256 + wave * 32;
    const int n0 = blockIdx.x * 64;

    const long arow0 = (long)imin(m0 + r, M - 1);
    const long arow1 = (long)imin(m0 + 16 + r, M - 1);

    // B-tile staging assignment: thread stages one 16-byte chunk of 64x32.
    const int srow   = tid >> 2;                       // 0..63
    const int schunk = tid & 3;                        // 0..3 (8 bf16 each)
    const long swrow = (long)imin(n0 + srow, N - 1);   // clamped W row
    bf16_t* ldsDst = &ldsB[srow * 32 + schunk * 8];

    const v8f vzero = {0.f, 0.f, 0.f, 0.f, 0.f, 0.f, 0.f, 0.f};
    AccU acc[2][4];
#pragma unroll
    for (int su = 0; su < 2; ++su)
#pragma unroll
        for (int t = 0; t < 4; ++t) acc[su][t].v = vzero;

    for (int k0 = 0; k0 < K; k0 += 32) {
        __syncthreads();  // previous iteration's LDS reads are done
        {
            const bf16_t* gsrc = Wb + swrow * (long)K + (long)(k0 + schunk * 8);
#if defined(ASYNC_LDS)
            __builtin_amdgcn_global_load_async_to_lds_b128(
                (v4i_t*)gsrc, (v4i_t*)ldsDst, 0, 0);
            __builtin_amdgcn_s_wait_asynccnt(0);
#else
            *(uint4*)ldsDst = *(const uint4*)gsrc;
#endif
        }
        __syncthreads();

#if defined(__gfx1250__)
        if (k0 + 32 < K)  // hint the next A slices into cache
            __builtin_prefetch((const void*)((const char*)Ab +
                               (arow0 * (long)K + (long)(k0 + 32)) * 2), 0, 1);
#endif
        FragBF a0 = load_frag(Ab, arow0, K, k0, koff);
        FragBF a1 = load_frag(Ab, arow1, K, k0, koff);
#pragma unroll
        for (int t = 0; t < 4; ++t) {
            FragBF b;
            const char* lp = (const char*)&ldsB[(t * 16 + r) * 32] + koff * 2;
            b.q[0] = *(const uint4*)lp;          // K = koff..koff+7
            b.q[1] = *(const uint4*)(lp + 32);   // K = 16+koff..+7
            acc[0][t].v = __builtin_amdgcn_wmma_f32_16x16x32_bf16(
                false, a0.v, false, b.v, (short)0, acc[0][t].v, false, false);
            acc[1][t].v = __builtin_amdgcn_wmma_f32_16x16x32_bf16(
                false, a1.v, false, b.v, (short)0, acc[1][t].v, false, false);
        }
    }

    // C/D layout: VGPR g, lanes 0-15 -> (M=g, N=lane); lanes 16-31 -> (M=8+g).
    const int mloc = hi * 8;
    const int ncol = lane & 15;
#pragma unroll
    for (int su = 0; su < 2; ++su) {
#pragma unroll
        for (int t = 0; t < 4; ++t) {
            const int n = n0 + t * 16 + ncol;
            if (n >= N) continue;
            const float bv = bias ? bias[n] : 0.f;
#pragma unroll
            for (int g = 0; g < 8; ++g) {
                const int m = m0 + su * 16 + mloc + g;
                if (m >= M) continue;
                float v = acc[su][t].f[g] + bv;
                if (ACT == 1)
                    v = 0.5f * v * (1.0f + erff(v * 0.70710678118654752f));
                const long oidx = obase + (long)m * oRowStride + n;
                if (OUTBF) ((bf16_t*)Out)[oidx] = (bf16_t)v;
                else       ((float*)Out)[oidx]  = v;
            }
        }
    }
}

// ---------------------------------------------------------------------------
// Elementwise / data-movement kernels
// ---------------------------------------------------------------------------
__global__ void cvt_bf16_kernel(const float* __restrict__ src,
                                bf16_t* __restrict__ dst, long n) {
    long i = (long)blockIdx.x * 256 + threadIdx.x;
    if (i < n) dst[i] = (bf16_t)src[i];
}

// 4x4 average pool -> global tokens tg[b][sg][c] (sg = y4*16 + x4), bf16.
__global__ void pool_tokens_kernel(const float* __restrict__ x,
                                   bf16_t* __restrict__ tg) {
    long idx = (long)blockIdx.x * 256 + threadIdx.x;
    const long total = (long)MG_ * C_;
    if (idx >= total) return;
    int c = (int)(idx % C_);
    long t = idx / C_;
    int sg = (int)(t % SG_);
    int b  = (int)(t / SG_);
    int y4 = sg >> 4, x4 = sg & 15;
    const float* xp = x + ((((long)b * C_ + c) * H_ + y4 * 4) * W_ + x4 * 4);
    float s = 0.f;
#pragma unroll
    for (int i = 0; i < 4; ++i)
#pragma unroll
        for (int j = 0; j < 4; ++j) s += xp[i * W_ + j];
    tg[idx] = (bf16_t)(s * 0.0625f);
}

// Window partition: xl[row][c], row = ((b*8+wy)*8+wx)*64 + iy*8+ix, bf16.
__global__ void win_tokens_kernel(const float* __restrict__ x,
                                  bf16_t* __restrict__ xl) {
    long idx = (long)blockIdx.x * 256 + threadIdx.x;
    const long total = (long)NTOK_ * C_;
    if (idx >= total) return;
    int c = (int)(idx % C_);
    long t = idx / C_;
    int inner = (int)(t % 64); long t2 = t / 64;
    int wx = (int)(t2 % 8);    long t3 = t2 / 8;
    int wy = (int)(t3 % 8);
    int b  = (int)(t3 / 8);
    int iy = inner >> 3, ix = inner & 7;
    int y = wy * WS_ + iy, xx = wx * WS_ + ix;
    xl[idx] = (bf16_t)x[(((long)b * C_ + c) * H_ + y) * W_ + xx];
}

// qkv (Mtok x 2304, bf16) -> q/k per (batch,head) [bh][s][hd] and vT [bh][hd][s]
__global__ __launch_bounds__(256) void split_qkv_kernel(
    const bf16_t* __restrict__ qkv, bf16_t* __restrict__ q,
    bf16_t* __restrict__ k, bf16_t* __restrict__ vT, int S, float qscale) {
    int m = blockIdx.x;
    int batch = m / S, s = m % S;
    long base = (long)m * (3 * C_);
    for (int c = threadIdx.x; c < C_; c += 256) {
        int h = c / HD_, d = c % HD_;
        float qv = (float)qkv[base + c] * qscale;
        float kv = (float)qkv[base + C_ + c];
        float vv = (float)qkv[base + 2 * C_ + c];
        long bh = (long)batch * HEADS_ + h;
        q[(bh * S + s) * HD_ + d] = (bf16_t)qv;
        k[(bh * S + s) * HD_ + d] = (bf16_t)kv;
        vT[(bh * HD_ + d) * S + s] = (bf16_t)vv;
    }
}

// Row softmax: scores f32 (rows x n) -> P bf16. One 256-thread block per row.
__global__ __launch_bounds__(256) void softmax_kernel(
    const float* __restrict__ S, bf16_t* __restrict__ P, int n) {
    long row = blockIdx.x;
    const float* sp = S + row * (long)n;
    int tid = threadIdx.x;
    __shared__ float red[8];

    float v = (tid < n) ? sp[tid] : -3.4e38f;
#pragma unroll
    for (int off = 16; off; off >>= 1) v = fmaxf(v, __shfl_xor(v, off, 32));
    if ((tid & 31) == 0) red[tid >> 5] = v;
    __syncthreads();
    float m = red[0];
#pragma unroll
    for (int i = 1; i < 8; ++i) m = fmaxf(m, red[i]);
    __syncthreads();

    float e = (tid < n) ? expf(sp[tid] - m) : 0.f;
    float s = e;
#pragma unroll
    for (int off = 16; off; off >>= 1) s += __shfl_xor(s, off, 32);
    if ((tid & 31) == 0) red[tid >> 5] = s;
    __syncthreads();
    float tot = 0.f;
#pragma unroll
    for (int i = 0; i < 8; ++i) tot += red[i];

    if (tid < n) P[row * (long)n + tid] = (bf16_t)(e / tot);
}

// ---- helpers for LN kernels ------------------------------------------------
static __device__ inline float block_sum256(float v, float* red) {
#pragma unroll
    for (int off = 16; off; off >>= 1) v += __shfl_xor(v, off, 32);
    int tid = threadIdx.x;
    __syncthreads();
    if ((tid & 31) == 0) red[tid >> 5] = v;
    __syncthreads();
    float s = 0.f;
#pragma unroll
    for (int i = 0; i < 8; ++i) s += red[i];
    return s;
}

static __device__ inline float cubw(float x) {  // bicubic a=-0.75
    float ax = fabsf(x);
    if (ax <= 1.f) return (1.25f * ax - 2.25f) * ax * ax + 1.f;
    if (ax < 2.f)  return ((-0.75f * ax + 3.75f) * ax - 6.0f) * ax + 3.0f;
    return 0.f;
}

// Fused: u = x + bicubic4(ga) + la ; t = LN1(u); write t (f32) and t (bf16).
// One block per token (b,s).
__global__ __launch_bounds__(256) void fuse_ln1_kernel(
    const float* __restrict__ x, const float* __restrict__ ga,
    const float* __restrict__ la, const float* __restrict__ g1,
    const float* __restrict__ bt1, float* __restrict__ tF,
    bf16_t* __restrict__ tB) {
    int token = blockIdx.x;
    int b = token >> 12, s = token & 4095;
    int y = s >> 6, xx = s & 63;

    float wy[4], wxv[4];
    int iy[4], ixv[4];
    {
        float src = (y + 0.5f) * 0.25f - 0.5f;
        int t0 = (int)floorf(src);
#pragma unroll
        for (int i = 0; i < 4; ++i) {
            int tp = t0 - 1 + i;
            wy[i] = cubw(src - (float)tp);
            iy[i] = imin(tp < 0 ? 0 : tp, 15);
        }
        src = (xx + 0.5f) * 0.25f - 0.5f;
        t0 = (int)floorf(src);
#pragma unroll
        for (int i = 0; i < 4; ++i) {
            int tp = t0 - 1 + i;
            wxv[i] = cubw(src - (float)tp);
            ixv[i] = imin(tp < 0 ? 0 : tp, 15);
        }
    }
    int wy_ = y >> 3, iy_ = y & 7, wx_ = xx >> 3, ix_ = xx & 7;
    long laRow = ((((long)b * 8 + wy_) * 8 + wx_) * 64 + iy_ * 8 + ix_) * C_;

    __shared__ float ubuf[C_];
    __shared__ float red[8];
    float pS = 0.f, pQ = 0.f;
    for (int c = threadIdx.x; c < C_; c += 256) {
        float xv = x[(((long)b * C_ + c) * H_ + y) * W_ + xx];
        const float* gb = ga + ((long)b * SG_) * C_ + c;
        float gv = 0.f;
#pragma unroll
        for (int i = 0; i < 4; ++i) {
            float rowAcc = 0.f;
#pragma unroll
            for (int j = 0; j < 4; ++j)
                rowAcc += wxv[j] * gb[(iy[i] * 16 + ixv[j]) * C_];
            gv += wy[i] * rowAcc;
        }
        float uv = xv + gv + la[laRow + c];
        ubuf[c] = uv;
        pS += uv; pQ += uv * uv;
    }
    float S = block_sum256(pS, red);
    float Q = block_sum256(pQ, red);
    float mean = S * (1.0f / C_);
    float var = Q * (1.0f / C_) - mean * mean;
    float rstd = rsqrtf(var + 1e-12f);
    for (int c = threadIdx.x; c < C_; c += 256) {
        float t = (ubuf[c] - mean) * rstd * g1[c] + bt1[c];
        long o = (long)token * C_ + c;
        tF[o] = t;
        tB[o] = (bf16_t)t;
    }
}

// Final: u = t + mlp ; out = LN2(u), permuted to (B,C,H,W).
__global__ __launch_bounds__(256) void final_ln2_kernel(
    const float* __restrict__ tF, const float* __restrict__ mlp,
    const float* __restrict__ g2, const float* __restrict__ bt2,
    float* __restrict__ out) {
    int token = blockIdx.x;
    int b = token >> 12, s = token & 4095;
    int y = s >> 6, xx = s & 63;
    __shared__ float ubuf[C_];
    __shared__ float red[8];
    float pS = 0.f, pQ = 0.f;
    for (int c = threadIdx.x; c < C_; c += 256) {
        long o = (long)token * C_ + c;
        float uv = tF[o] + mlp[o];
        ubuf[c] = uv;
        pS += uv; pQ += uv * uv;
    }
    float S = block_sum256(pS, red);
    float Q = block_sum256(pQ, red);
    float mean = S * (1.0f / C_);
    float rstd = rsqrtf(Q * (1.0f / C_) - mean * mean + 1e-12f);
    for (int c = threadIdx.x; c < C_; c += 256) {
        out[(((long)b * C_ + c) * H_ + y) * W_ + xx] =
            (ubuf[c] - mean) * rstd * g2[c] + bt2[c];
    }
}

// ---------------------------------------------------------------------------
// Host orchestration
// ---------------------------------------------------------------------------
static inline size_t aln(size_t x) { return (x + 255) & ~(size_t)255; }
static inline unsigned cdiv(long a, long b) { return (unsigned)((a + b - 1) / b); }

extern "C" void kernel_launch(void* const* d_in, const int* in_sizes, int n_in,
                              void* d_out, int out_size, void* d_ws, size_t ws_size,
                              hipStream_t stream) {
    (void)in_sizes; (void)n_in; (void)out_size;
    const float* x       = (const float*)d_in[0];
    const float* w_in_g  = (const float*)d_in[2];
    const float* b_in_g  = (const float*)d_in[3];
    const float* w_out_g = (const float*)d_in[4];
    const float* b_out_g = (const float*)d_in[5];
    const float* w_in_l  = (const float*)d_in[6];
    const float* b_in_l  = (const float*)d_in[7];
    const float* w_out_l = (const float*)d_in[8];
    const float* b_out_l = (const float*)d_in[9];
    const float* ln1_g   = (const float*)d_in[10];
    const float* ln1_b   = (const float*)d_in[11];
    const float* ln2_g   = (const float*)d_in[12];
    const float* ln2_b   = (const float*)d_in[13];
    const float* w1      = (const float*)d_in[14];
    const float* b1      = (const float*)d_in[15];
    const float* w2      = (const float*)d_in[16];
    const float* b2      = (const float*)d_in[17];
    float* out = (float*)d_out;
    unsigned char* ws = (unsigned char*)d_ws;

    const size_t MB = 1ull << 20;
    // ---- persistent allocations ----
    size_t off = 0;
    auto ALLOC = [&](size_t bytes) { size_t o = off; off = aln(off + bytes); return o; };
    size_t o_wing  = ALLOC((size_t)3 * C_ * C_ * 2);       // w_in_g  bf16 (2304x768)
    size_t o_woutg = ALLOC((size_t)C_ * C_ * 2);           // w_out_g bf16
    size_t o_winl  = ALLOC((size_t)3 * C_ * C_ * 2);
    size_t o_woutl = ALLOC((size_t)C_ * C_ * 2);
    size_t o_w1    = ALLOC((size_t)HID_ * C_ * 2);
    size_t o_w2    = ALLOC((size_t)C_ * HID_ * 2);
    size_t o_ga    = ALLOC((size_t)MG_ * C_ * 4);          // global-branch tokens f32
    size_t o_la    = ALLOC((size_t)NTOK_ * C_ * 4);        // local-branch tokens f32
    size_t o_tF    = ALLOC((size_t)NTOK_ * C_ * 4);        // post-LN1 f32
    size_t o_tB    = ALLOC((size_t)NTOK_ * C_ * 2);        // post-LN1 bf16
    size_t o_xl    = ALLOC((size_t)NTOK_ * C_ * 2);        // window tokens bf16
    size_t scr = off;
    // ---- scratch (phase-overlapped) ----
    // Phase G:
    size_t o_tg   = scr + 0;        // 3.1 MB
    size_t o_qkvg = scr + 4 * MB;   // 9.4 MB
    size_t o_qg   = scr + 16 * MB;  // 3.15 MB each
    size_t o_kg   = scr + 20 * MB;
    size_t o_vgT  = scr + 24 * MB;
    size_t o_sg   = scr + 28 * MB;  // 16.8 MB f32
    size_t o_pg   = scr + 46 * MB;  // 8.4 MB bf16
    size_t o_og   = scr + 56 * MB;  // 3.1 MB bf16
    // Phase L (reuses phase-G region):
    size_t o_qkvl = scr + 0;        // 151 MB bf16
    size_t o_ql   = scr + 152 * MB; // 50.4 MB each
    size_t o_kl   = scr + 203 * MB;
    size_t o_vlT  = scr + 254 * MB;
    size_t o_sl   = scr + 0;        // 67 MB f32  (qkvl dead)
    size_t o_pl   = scr + 68 * MB;  // 33.6 MB bf16
    size_t o_ol   = scr + 0;        // 50.4 MB bf16 (scores dead; ql/kl dead)
    // Phase M:
    size_t o_hid  = scr + 0;        // 201.3 MB bf16
    size_t o_mlp  = scr + 202 * MB; // 100.7 MB f32
    size_t total = scr + 305 * MB;
    if (ws_size < total) return;  // insufficient scratch

    const float qscale = 0.1020620726159658f;  // 1/sqrt(96)

    auto cvt = [&](const float* src, size_t dst, long n) {
        cvt_bf16_kernel<<<dim3(cdiv(n, 256)), dim3(256), 0, stream>>>(
            src, (bf16_t*)(ws + dst), n);
    };
    cvt(w_in_g, o_wing, (long)3 * C_ * C_);
    cvt(w_out_g, o_woutg, (long)C_ * C_);
    cvt(w_in_l, o_winl, (long)3 * C_ * C_);
    cvt(w_out_l, o_woutl, (long)C_ * C_);
    cvt(w1, o_w1, (long)HID_ * C_);
    cvt(w2, o_w2, (long)C_ * HID_);

    pool_tokens_kernel<<<dim3(cdiv((long)MG_ * C_, 256)), dim3(256), 0, stream>>>(
        x, (bf16_t*)(ws + o_tg));
    win_tokens_kernel<<<dim3(cdiv((long)NTOK_ * C_, 256)), dim3(256), 0, stream>>>(
        x, (bf16_t*)(ws + o_xl));

    // ---------------- global branch ----------------
    gemm_wmma_kernel<0, 1><<<dim3(36, 8, 1), 256, 0, stream>>>(
        (const bf16_t*)(ws + o_tg), (const bf16_t*)(ws + o_wing), b_in_g,
        (char*)(ws + o_qkvg), MG_, 3 * C_, C_, 0, 0, 1, 0, 0, 3 * C_);
    split_qkv_kernel<<<MG_, 256, 0, stream>>>(
        (const bf16_t*)(ws + o_qkvg), (bf16_t*)(ws + o_qg),
        (bf16_t*)(ws + o_kg), (bf16_t*)(ws + o_vgT), SG_, qscale);
    gemm_wmma_kernel<0, 0><<<dim3(4, 1, B_ * HEADS_), 256, 0, stream>>>(
        (const bf16_t*)(ws + o_qg), (const bf16_t*)(ws + o_kg), nullptr,
        (char*)(ws + o_sg), SG_, SG_, HD_, (long)SG_ * HD_, (long)SG_ * HD_, 1,
        (long)SG_ * SG_, 0, SG_);
    softmax_kernel<<<B_ * HEADS_ * SG_, 256, 0, stream>>>(
        (const float*)(ws + o_sg), (bf16_t*)(ws + o_pg), SG_);
    gemm_wmma_kernel<0, 1><<<dim3(2, 1, B_ * HEADS_), 256, 0, stream>>>(
        (const bf16_t*)(ws + o_pg), (const bf16_t*)(ws + o_vgT), nullptr,
        (char*)(ws + o_og), SG_, HD_, SG_, (long)SG_ * SG_, (long)HD_ * SG_,
        HEADS_, (long)SG_ * C_, HD_, C_);
    gemm_wmma_kernel<0, 0><<<dim3(12, 8, 1), 256, 0, stream>>>(
        (const bf16_t*)(ws + o_og), (const bf16_t*)(ws + o_woutg), b_out_g,
        (char*)(ws + o_ga), MG_, C_, C_, 0, 0, 1, 0, 0, C_);

    // ---------------- local branch ----------------
    gemm_wmma_kernel<0, 1><<<dim3(36, 128, 1), 256, 0, stream>>>(
        (const bf16_t*)(ws + o_xl), (const bf16_t*)(ws + o_winl), b_in_l,
        (char*)(ws + o_qkvl), NTOK_, 3 * C_, C_, 0, 0, 1, 0, 0, 3 * C_);
    split_qkv_kernel<<<NTOK_, 256, 0, stream>>>(
        (const bf16_t*)(ws + o_qkvl), (bf16_t*)(ws + o_ql),
        (bf16_t*)(ws + o_kl), (bf16_t*)(ws + o_vlT), 64, qscale);
    gemm_wmma_kernel<0, 0><<<dim3(1, 1, NWIN_ * HEADS_), 256, 0, stream>>>(
        (const bf16_t*)(ws + o_ql), (const bf16_t*)(ws + o_kl), nullptr,
        (char*)(ws + o_sl), 64, 64, HD_, (long)64 * HD_, (long)64 * HD_, 1,
        (long)64 * 64, 0, 64);
    softmax_kernel<<<NWIN_ * HEADS_ * 64, 256, 0, stream>>>(
        (const float*)(ws + o_sl), (bf16_t*)(ws + o_pl), 64);
    gemm_wmma_kernel<0, 1><<<dim3(2, 1, NWIN_ * HEADS_), 256, 0, stream>>>(
        (const bf16_t*)(ws + o_pl), (const bf16_t*)(ws + o_vlT), nullptr,
        (char*)(ws + o_ol), 64, HD_, 64, (long)64 * 64, (long)HD_ * 64,
        HEADS_, (long)64 * C_, HD_, C_);
    gemm_wmma_kernel<0, 0><<<dim3(12, 128, 1), 256, 0, stream>>>(
        (const bf16_t*)(ws + o_ol), (const bf16_t*)(ws + o_woutl), b_out_l,
        (char*)(ws + o_la), NTOK_, C_, C_, 0, 0, 1, 0, 0, C_);

    // ---------------- fuse + LN1, MLP, LN2 ----------------
    fuse_ln1_kernel<<<NTOK_, 256, 0, stream>>>(
        x, (const float*)(ws + o_ga), (const float*)(ws + o_la), ln1_g, ln1_b,
        (float*)(ws + o_tF), (bf16_t*)(ws + o_tB));
    gemm_wmma_kernel<1, 1><<<dim3(48, 128, 1), 256, 0, stream>>>(
        (const bf16_t*)(ws + o_tB), (const bf16_t*)(ws + o_w1), b1,
        (char*)(ws + o_hid), NTOK_, HID_, C_, 0, 0, 1, 0, 0, HID_);
    gemm_wmma_kernel<0, 0><<<dim3(12, 128, 1), 256, 0, stream>>>(
        (const bf16_t*)(ws + o_hid), (const bf16_t*)(ws + o_w2), b2,
        (char*)(ws + o_mlp), NTOK_, C_, HID_, 0, 0, 1, 0, 0, C_);
    final_ln2_kernel<<<NTOK_, 256, 0, stream>>>(
        (const float*)(ws + o_tF), (const float*)(ws + o_mlp), ln2_g, ln2_b, out);
}
